// PatchMatch_86955907875355
// MI455X (gfx1250) — matrix operand order repaired and checked by reading, not compiled, not run on hardware
//
#include <hip/hip_runtime.h>
#include <hip/hip_bf16.h>
#include <math.h>

typedef __attribute__((ext_vector_type(16))) _Float16 v16h;
typedef __attribute__((ext_vector_type(8)))  float    v8f;

#define BB 4
#define GG 8
#define DD 8
#define HH 256
#define WW 320
#define NN 9
#define HW (HH*WW)          // 81920
#define DHW (DD*HH*WW)      // 655360
#define TOTAL (BB*DHW)      // 2621440
#define TILES (TOTAL/16)    // 163840
#define BN_EPS 1e-5f

// Workspace: float2 field[TOTAL]; field[i].x = xnorm, field[i].y = s.
// Interleaving makes every bilinear corner gather in agg_kernel one b64 load.

// ---------------- Kernel 1: xnorm = (1/ds - inv_max) / (inv_min - inv_max) ----------------
__global__ void xnorm_kernel(const float* __restrict__ ds,
                             const float* __restrict__ dmin,
                             const float* __restrict__ dmax,
                             float* __restrict__ field) {   // writes .x slots (stride 2)
    int i = blockIdx.x * blockDim.x + threadIdx.x;
    if (i >= TOTAL) return;
    int b = i / DHW;
    float inv_min = 1.0f / dmin[b];
    float inv_max = 1.0f / dmax[b];
    field[2 * i] = (1.0f / ds[i] - inv_max) / (inv_min - inv_max);
}

// ---------------- Kernel 2: SimilarityNet via WMMA (G->16->8->1, folded BN+ReLU) ----------
// Weights in A (M = out channels), 16 voxels in B's N dim. BN shift rides in C-init,
// BN scale folded into A rows. Layer-2 K-mapping: channels 0..7 -> K=0..7,
// channels 8..15 -> K=16..23, so D1 feeds B2 with NO cross-lane shuffle:
// B2 in contiguous K order: lane n elems0..7 = K0..7 and lane n+16 elems0..7 = K16..23,
// which is exactly where D1's (ch0..7 / ch8..15) data already lives.
__global__ void sim_kernel(const float* __restrict__ cost,
                           const float* __restrict__ w0,
                           const float* __restrict__ g0, const float* __restrict__ b0,
                           const float* __restrict__ m0, const float* __restrict__ v0,
                           const float* __restrict__ w1,
                           const float* __restrict__ g1, const float* __restrict__ b1n,
                           const float* __restrict__ m1, const float* __restrict__ v1,
                           const float* __restrict__ w2, const float* __restrict__ b2,
                           float* __restrict__ field) {     // writes .y slots (stride 2)
    const int lane  = threadIdx.x & 31;
    const int wave  = blockIdx.x * (blockDim.x >> 5) + (threadIdx.x >> 5);
    const int nwav  = gridDim.x * (blockDim.x >> 5);
    const bool lo   = lane < 16;
    const int  n    = lane & 15;

    // ---- per-wave constant setup ----
    // A1 (16x32 f16 A-layout): lane m elems0..7 = K0..7 = w0[m][*]*scale0 ; everything else 0.
    v16h a1, a2;
    {
        float s0 = g0[n] * rsqrtf(v0[n] + BN_EPS);
        #pragma unroll
        for (int g = 0; g < 8; ++g) {
            float wv = w0[n * GG + g] * s0;
            a1[g]     = lo ? (_Float16)wv : (_Float16)0.0f;   // hi lanes = K8..15 -> 0
            a1[8 + g] = (_Float16)0.0f;                       // K16..23 / K24..31 -> 0
        }
    }
    {
        // A2: row m (m<8): lo lanes elems0..7 = w1[m][0..7]*s1, hi lanes elems0..7 = w1[m][8..15]*s1, rest 0.
        int   mq = n & 7;
        bool  rowok = (n < 8);
        float s1 = g1[mq] * rsqrtf(v1[mq] + BN_EPS);
        #pragma unroll
        for (int j = 0; j < 8; ++j) {
            float wlo = w1[mq * 16 + (lo ? 0 : 8) + j] * s1;   // ch j (lo lanes) / ch 8+j (hi lanes), elems 0..7
            a2[j]     = rowok ? (_Float16)wlo : (_Float16)0.0f;
            a2[8 + j] = (_Float16)0.0f;
        }
    }
    // C init = BN shift. C layout: vgpr r holds row M = r + 8*(lane>=16).
    v8f c1, c2;
    #pragma unroll
    for (int r = 0; r < 8; ++r) {
        int m = r + (lo ? 0 : 8);
        float s0 = g0[m] * rsqrtf(v0[m] + BN_EPS);
        c1[r] = b0[m] - m0[m] * s0;
        int   mq = m & 7;
        float s1 = g1[mq] * rsqrtf(v1[mq] + BN_EPS);
        c2[r] = (m < 8) ? (b1n[mq] - m1[mq] * s1) : 0.0f;
    }
    float w2r[8];
    #pragma unroll
    for (int i = 0; i < 8; ++i) w2r[i] = w2[i];
    const float bias2 = b2[0];

    // ---- main loop: one 16-voxel tile per wave-iteration, 2 WMMAs, no shuffles ----
    for (int t = wave; t < TILES; t += nwav) {
        int v = t * 16 + n;                 // hi lanes mirror lo lanes' addresses
        int b = v / DHW;
        int r = v - b * DHW;
        const float* cb = cost + (size_t)b * (GG * DHW) + r;

        // B1: elems0..7 = cost g=0..7 (garbage in hi lanes is multiplied by A1's zeros).
        v16h b1v;
        #pragma unroll
        for (int g = 0; g < 8; ++g) {
            b1v[g]     = (_Float16)cb[(size_t)g * DHW];   // coalesced across voxel lanes
            b1v[8 + g] = (_Float16)0.0f;
        }
        v8f d1 = __builtin_amdgcn_wmma_f32_16x16x32_f16(
            false, a1, false, b1v, (short)0, c1, false, false);

        // B2 = relu(d1) in place: lo lanes supply K0..7 (ch0..7), hi lanes K16..23 (ch8..15).
        v16h b2v;
        #pragma unroll
        for (int i = 0; i < 8; ++i) {
            b2v[i]     = (_Float16)fmaxf(d1[i], 0.0f);
            b2v[8 + i] = (_Float16)0.0f;
        }
        v8f d2 = __builtin_amdgcn_wmma_f32_16x16x32_f16(
            false, a2, false, b2v, (short)0, c2, false, false);

        // Head: s = b2 + sum_c w2[c]*relu(h2[c]); lane n (lo) owns voxel n's ch0..7.
        float sv = bias2;
        #pragma unroll
        for (int i = 0; i < 8; ++i) sv += w2r[i] * fmaxf(d2[i], 0.0f);
        if (lo) field[2 * (b * DHW + r) + 1] = sv;
    }
}

// ---------------- Kernel 3: 9-neighbor bilinear sample + sigmoid weight + reduce ----------
__global__ void agg_kernel(const float2* __restrict__ field,
                           const float* __restrict__ grid,
                           const float* __restrict__ fw,
                           float* __restrict__ out) {
    int idx = blockIdx.x * blockDim.x + threadIdx.x;   // over B*H*W
    if (idx >= BB * HW) return;
    int w = idx % WW;
    int h = (idx / WW) % HH;
    int b = idx / HW;

    int base = b * DHW;
    int pix  = h * WW + w;
    float xc[DD], acc[DD];
    #pragma unroll
    for (int d = 0; d < DD; ++d) {
        xc[d]  = field[base + d * HW + pix].x;
        acc[d] = 0.0f;
    }

    #pragma unroll 1
    for (int nn = 0; nn < NN; ++nn) {
        int gidx = (((b * NN + nn) * HH + h) * WW + w) * 2;
        float gxn = grid[gidx];
        float gyn = grid[gidx + 1];
        float gx = fminf(fmaxf((gxn + 1.0f) * 0.5f * WW - 0.5f, 0.0f), (float)(WW - 1));
        float gy = fminf(fmaxf((gyn + 1.0f) * 0.5f * HH - 0.5f, 0.0f), (float)(HH - 1));
        float x0f = floorf(gx), y0f = floorf(gy);
        int x0 = (int)x0f, y0 = (int)y0f;
        int x1 = min(x0 + 1, WW - 1), y1 = min(y0 + 1, HH - 1);
        float wx = gx - x0f, wy = gy - y0f;
        float w00 = (1.0f - wx) * (1.0f - wy);
        float w01 = wx * (1.0f - wy);
        float w10 = (1.0f - wx) * wy;
        float w11 = wx * wy;
        int o00 = y0 * WW + x0, o01 = y0 * WW + x1;
        int o10 = y1 * WW + x0, o11 = y1 * WW + x1;
        float fwv = fw[((b * NN + nn) * HH + h) * WW + w];

        #pragma unroll
        for (int d = 0; d < DD; ++d) {
            int pb = base + d * HW;
            float2 q00 = field[pb + o00];
            float2 q01 = field[pb + o01];
            float2 q10 = field[pb + o10];
            float2 q11 = field[pb + o11];
            float xs = q00.x * w00 + q01.x * w01 + q10.x * w10 + q11.x * w11;
            float ss = q00.y * w00 + q01.y * w01 + q10.y * w10 + q11.y * w11;
            float diff = fminf(fabsf(xs - xc[d]) * 40.0f, 4.0f);   // /0.025, clip [0,4]
            float dw = 1.0f / (1.0f + __expf(-(2.0f - diff) * 2.0f));
            acc[d] += ss * fwv * dw;
        }
    }
    #pragma unroll
    for (int d = 0; d < DD; ++d)
        out[base + d * HW + pix] = acc[d];
}

extern "C" void kernel_launch(void* const* d_in, const int* in_sizes, int n_in,
                              void* d_out, int out_size, void* d_ws, size_t ws_size,
                              hipStream_t stream) {
    const float* cost = (const float*)d_in[0];
    const float* ds   = (const float*)d_in[1];
    const float* dmin = (const float*)d_in[2];
    const float* dmax = (const float*)d_in[3];
    const float* grid = (const float*)d_in[4];
    const float* fw   = (const float*)d_in[5];
    const float* w0   = (const float*)d_in[6];
    const float* g0   = (const float*)d_in[7];
    const float* b0   = (const float*)d_in[8];
    const float* m0   = (const float*)d_in[9];
    const float* v0   = (const float*)d_in[10];
    const float* w1   = (const float*)d_in[11];
    const float* g1   = (const float*)d_in[12];
    const float* b1   = (const float*)d_in[13];
    const float* m1   = (const float*)d_in[14];
    const float* v1   = (const float*)d_in[15];
    const float* w2   = (const float*)d_in[16];
    const float* b2   = (const float*)d_in[17];
    float* out   = (float*)d_out;
    float* fieldf = (float*)d_ws;                 // float2[TOTAL] interleaved {xnorm, s}

    xnorm_kernel<<<(TOTAL + 255) / 256, 256, 0, stream>>>(ds, dmin, dmax, fieldf);
    // 2048 blocks * 8 waves = 16384 waves, ~10 tiles each (grid-stride)
    sim_kernel<<<2048, 256, 0, stream>>>(cost, w0, g0, b0, m0, v0,
                                         w1, g1, b1, m1, v1, w2, b2, fieldf);
    agg_kernel<<<(BB * HW + 255) / 256, 256, 0, stream>>>((const float2*)fieldf,
                                                          grid, fw, out);
}
